// IntConv2d_42992622633269
// MI455X (gfx1250) — 8-bit matrix operand order repaired and checked by reading, compiled, not run on hardware
//
#include <hip/hip_runtime.h>

typedef __attribute__((ext_vector_type(8))) int v8i;
typedef __attribute__((ext_vector_type(4))) int v4i;
typedef unsigned long long ull;

#define NB   16
#define HH_  112
#define WW_  112
#define HP_  114
#define WP_  114
#define CIN_ 256
#define COUT_ 256

__global__ void pack_x_pad_kernel(const int* __restrict__ x, int* __restrict__ xp, int n4) {
    int i = blockIdx.x * blockDim.x + threadIdx.x;
    if (i >= n4) return;
    int c4 = i & 63;
    int t  = i >> 6;
    int wp = t % WP_;
    int hp = (t / WP_) % HP_;
    int nn = t / (WP_ * HP_);
    int h = hp - 1, w = wp - 1;
    int packed = 0;
    if (h >= 0 && h < HH_ && w >= 0 && w < WW_) {
        const v4i* src = (const v4i*)x + (((size_t)nn * HH_ + h) * WW_ + w) * 64 + c4;
        const v4i v = __builtin_nontemporal_load(src);
        packed = (v.x & 0xff) | ((v.y & 0xff) << 8) | ((v.z & 0xff) << 16) | ((v.w & 0xff) << 24);
    }
    xp[i] = packed;
}

__global__ void pack_w_kernel(const int* __restrict__ w, signed char* __restrict__ wi8) {
    int idx = blockIdx.x * blockDim.x + threadIdx.x;
    if (idx >= COUT_ * 3 * 3 * CIN_) return;
    int b     = idx & 31;
    int lane  = (idx >> 5) & 31;
    int kc    = (idx >> 10) % 36;
    int ntile = idx / (36 * 1024);
    int nloc  = lane & 15;
    int hi    = lane >> 4;
    int seg   = b >> 4;
    int off   = b & 15;
    int klocal = (b >> 3) * 16 + hi * 8 + (b & 7);
    int khw = kc >> 2;
    int cb  = kc & 3;
    int kh  = khw / 3;
    int kw  = khw - kh * 3;
    int cin  = cb * 64 + klocal;
    int cout = ntile * 16 + nloc;
    int src = ((cout * 3 + kh) * 3 + kw) * CIN_ + cin;
    wi8[idx] = (signed char)w[src];
}

#define WNT_BYTES 36864
#define SMEM_BYTES (4 * WNT_BYTES)
#define ROWB (WP_ * 256)

__global__ void __launch_bounds__(256)
conv_wmma_kernel(const signed char* __restrict__ xpad,
                 const signed char* __restrict__ wi8,
                 int* __restrict__ out) {
    extern __shared__ signed char smem[];

    const int wave = threadIdx.x >> 5;
    const int lane = threadIdx.x & 31;

    const int ngrp  = blockIdx.x & 3;
    const int mtile = (blockIdx.x >> 2) * 8 + wave;
    const int wt = mtile % 7;
    const int hh = (mtile / 7) % HH_;
    const int nn = mtile / (7 * HH_);
    const int w0 = wt * 16;

    const int m   = lane & 15;
    const int hi8 = (lane >> 4) << 3;

    {
        const v4i* src = (const v4i*)(wi8 + (size_t)(ngrp * 4) * WNT_BYTES);
        v4i* dst = (v4i*)smem;
        #pragma unroll
        for (int g = 0; g < 3; ++g) {
            v4i t[12];
            #pragma unroll
            for (int i = 0; i < 12; ++i)
                t[i] = src[(g * 12 + i) * 256 + threadIdx.x];
            #pragma unroll
            for (int i = 0; i < 12; ++i)
                dst[(g * 12 + i) * 256 + threadIdx.x] = t[i];
        }
    }
    __syncthreads();

    v8i acc0 = {}, acc1 = {}, acc2 = {}, acc3 = {};

    const signed char* abase =
        xpad + (((size_t)nn * HP_ + hh) * WP_ + (w0 + m)) * 256 + hi8;
    const signed char* sfrag = smem + lane * 32;

    #pragma unroll
    for (int kh = 0; kh < 3; ++kh) {
        #pragma unroll
        for (int kw = 0; kw < 3; ++kw) {
            const signed char* ap = abase + kh * ROWB + kw * 256;
            const signed char* bp = sfrag + (kh * 3 + kw) * 4096;
            #pragma unroll
            for (int cb = 0; cb < 4; ++cb) {
                ull q0 = *(const ull*)(ap + cb * 64 + 0);
                ull q1 = *(const ull*)(ap + cb * 64 + 16);
                ull q2 = *(const ull*)(ap + cb * 64 + 32);
                ull q3 = *(const ull*)(ap + cb * 64 + 48);
                v8i a;
                a[0] = (int)q0; a[1] = (int)(q0 >> 32);
                a[2] = (int)q1; a[3] = (int)(q1 >> 32);
                a[4] = (int)q2; a[5] = (int)(q2 >> 32);
                a[6] = (int)q3; a[7] = (int)(q3 >> 32);

                v8i b0 = *(const v8i*)(bp + cb * 1024 + 0 * WNT_BYTES);
                v8i b1 = *(const v8i*)(bp + cb * 1024 + 1 * WNT_BYTES);
                v8i b2 = *(const v8i*)(bp + cb * 1024 + 2 * WNT_BYTES);
                v8i b3 = *(const v8i*)(bp + cb * 1024 + 3 * WNT_BYTES);

                acc0 = __builtin_amdgcn_wmma_i32_16x16x64_iu8(true, a, true, b0, acc0, false, false);
                acc1 = __builtin_amdgcn_wmma_i32_16x16x64_iu8(true, a, true, b1, acc1, false, false);
                acc2 = __builtin_amdgcn_wmma_i32_16x16x64_iu8(true, a, true, b2, acc2, false, false);
                acc3 = __builtin_amdgcn_wmma_i32_16x16x64_iu8(true, a, true, b3, acc3, false, false);
            }
        }
    }

    const int ncol  = ngrp * 64 + (lane & 15);
    const int mbase = (lane >> 4) << 3;
    int* obase = out + ((((size_t)nn * HH_ + hh) * WW_ + w0) + mbase) * COUT_ + ncol;
    #pragma unroll
    for (int j = 0; j < 8; ++j) {
        int* orow = obase + (size_t)j * COUT_;
        __builtin_nontemporal_store(acc0[j], orow + 0);
        __builtin_nontemporal_store(acc1[j], orow + 16);
        __builtin_nontemporal_store(acc2[j], orow + 32);
        __builtin_nontemporal_store(acc3[j], orow + 48);
    }
}

extern "C" void kernel_launch(void* const* d_in, const int* in_sizes, int n_in,
                              void* d_out, int out_size, void* d_ws, size_t ws_size,
                              hipStream_t stream) {
    const int* x = (const int*)d_in[0];
    const int* w = (const int*)d_in[1];
    int* out = (int*)d_out;

    signed char* xpad = (signed char*)d_ws;
    signed char* wi8  = xpad + (size_t)NB * HP_ * WP_ * CIN_;

    const int n4p = NB * HP_ * WP_ * (CIN_ / 4);
    pack_x_pad_kernel<<<(n4p + 255) / 256, 256, 0, stream>>>(x, (int*)xpad, n4p);

    const int wbytes = COUT_ * 3 * 3 * CIN_;
    pack_w_kernel<<<(wbytes + 255) / 256, 256, 0, stream>>>(w, wi8);

    const int nblocks = 4 * (NB * HH_ * (WW_ / 16) / 8);
    conv_wmma_kernel<<<nblocks, 256, SMEM_BYTES, stream>>>(xpad, wi8, out);
}
